// H_Softmax_40510131536001
// MI455X (gfx1250) — compile-verified, matrix operand order checked by reading, not run on hardware
//
#include <hip/hip_runtime.h>
#include <hip/hip_bf16.h>

// ---------------------------------------------------------------------------
// Hierarchical softmax for MI455X (gfx1250, wave32, WMMA).
//
//   delta[b,n] = x[b,:] . (W[n,:,1]-W[n,:,0]) + (b[n,1]-b[n,0])
//   s[b,n]     = sigmoid(delta[b,n])                (prob of bit==1 at node n)
//   out[b,leaf]= prod_d ( bit_d ? s[b,node_d] : 1 - s[b,node_d] )
//
// GEMM: M=256 (batch) x N=16384 (nodes, padded) x K=512, f16 WMMA 16x16x32,
// f32 accumulate. W (67MB) is read exactly once -> HBM floor ~2.9us.
// Double-buffered LDS staging, 1 barrier per K-step, 4 WMMAs per wave-step.
// ---------------------------------------------------------------------------

#define BATCH   256
#define EDIM    512
#define NODES   16383
#define NPAD    16384
#define VLEAF   16384
#define DEPTH   14

#define NT      32          // nodes per workgroup tile
#define LDS_STRIDE 40       // padded row stride in halfs (80B, 16B-aligned)

typedef __attribute__((ext_vector_type(16))) _Float16 v16h;
typedef __attribute__((ext_vector_type(8)))  _Float16 v8h;
typedef __attribute__((ext_vector_type(4)))  _Float16 v4h;
typedef __attribute__((ext_vector_type(8)))  float    v8f;

union V16H { v16h v; struct { v8h lo, hi; } s; };

// ---------------------------------------------------------------------------
// Kernel 1: convert x (f32, 256x512) -> f16 row-major table in workspace.
// ---------------------------------------------------------------------------
__global__ __launch_bounds__(256) void cvt_x_kernel(const float* __restrict__ x,
                                                    _Float16* __restrict__ xh) {
    int i = blockIdx.x * 256 + threadIdx.x;        // 131072 elements
    xh[i] = (_Float16)x[i];
}

// ---------------------------------------------------------------------------
// Kernel 2: GEMM (delta = x . (W1-W0)) + bias-delta + sigmoid -> s[b][n].
// Grid: 512 blocks (32 nodes each), 256 threads (8 waves).
// Wave w owns M rows [32w, 32w+32) x 32 nodes: 4 accumulators, 4 WMMAs/step.
// K loop: 16 steps of 32, double-buffered LDS, software-pipelined W loads.
// ---------------------------------------------------------------------------
__global__ __launch_bounds__(256) void gemm_sigmoid_kernel(
        const float* __restrict__ W,      // (NODES, EDIM, 2)
        const float* __restrict__ bias,   // (NODES, 2)
        const _Float16* __restrict__ xh,  // (BATCH, EDIM) f16
        float* __restrict__ s) {          // (BATCH, NPAD) f32
    __shared__ __align__(16) _Float16 ldsB[2][NT * LDS_STRIDE];   // 2 x 2.5KB

    const int nTile = blockIdx.x * NT;
    const int lane  = threadIdx.x & 31;
    const int wave  = threadIdx.x >> 5;   // 0..7
    const int mBase = wave * 32;

    // ISA 16-bit A layout (16x32): lanes 0-15 carry M=lane, K={0..7,16..23};
    // lanes 16-31 carry M=lane-16, K={8..15,24..31}. B mirrors over columns.
    const int mRow0 = mBase + (lane & 15);
    const int mRow1 = mRow0 + 16;
    const int kSel  = (lane >> 4) * 8;    // 0 or 8 (half offset inside K-chunk)
    const int nCol  = (lane & 15);

    // Staging assignment: thread -> (node nStg, 4 e-values at eStg).
    const int nStg  = threadIdx.x >> 3;          // 0..31
    const int eStg  = (threadIdx.x & 7) * 4;     // 0,4,...,28
    int nodeStg = nTile + nStg;
    nodeStg = nodeStg < NODES ? nodeStg : (NODES - 1);
    const float* __restrict__ wStg = W + ((size_t)nodeStg * EDIM + eStg) * 2;

    v8f acc[4] = {{}, {}, {}, {}};

    // Prologue: load first W tile chunk (8 f32 = two float4 per thread).
    float4 r0 = *(const float4*)(wStg + 0);
    float4 r1 = *(const float4*)(wStg + 4);

    int buf = 0;
    for (int k0 = 0; k0 < EDIM; k0 += 32) {
        // Store staged delta weights (4 halfs) into current LDS buffer.
        {
            v4h pk;
            pk[0] = (_Float16)(r0.y - r0.x);
            pk[1] = (_Float16)(r0.w - r0.z);
            pk[2] = (_Float16)(r1.y - r1.x);
            pk[3] = (_Float16)(r1.w - r1.z);
            *(v4h*)&ldsB[buf][nStg * LDS_STRIDE + eStg] = pk;
        }
        // Prefetch next K-chunk of W into registers (overlaps with WMMAs).
        if (k0 + 32 < EDIM) {
            const float* wNext = wStg + (size_t)(k0 + 32) * 2;
            r0 = *(const float4*)(wNext + 0);
            r1 = *(const float4*)(wNext + 4);
        }
        __syncthreads();   // ldsB[buf] now valid for all waves

        // A fragments from global f16 x-table (L2 resident).
        V16H a0, a1;
        {
            const _Float16* p0 = xh + (size_t)mRow0 * EDIM + k0 + kSel;
            a0.s.lo = *(const v8h*)(p0);
            a0.s.hi = *(const v8h*)(p0 + 16);
            const _Float16* p1 = xh + (size_t)mRow1 * EDIM + k0 + kSel;
            a1.s.lo = *(const v8h*)(p1);
            a1.s.hi = *(const v8h*)(p1 + 16);
        }
        // Two B fragments (node columns nCol and nCol+16) + 4 WMMAs.
        #pragma unroll
        for (int j = 0; j < 2; ++j) {
            V16H bm;
            const _Float16* pb = &ldsB[buf][(nCol + 16 * j) * LDS_STRIDE + kSel];
            bm.s.lo = *(const v8h*)(pb);
            bm.s.hi = *(const v8h*)(pb + 16);
            acc[2 * j + 0] = __builtin_amdgcn_wmma_f32_16x16x32_f16(
                false, a0.v, false, bm.v, (short)0, acc[2 * j + 0], false, false);
            acc[2 * j + 1] = __builtin_amdgcn_wmma_f32_16x16x32_f16(
                false, a1.v, false, bm.v, (short)0, acc[2 * j + 1], false, false);
        }
        buf ^= 1;
    }

    // Epilogue: add bias delta, sigmoid, store s[b][n].
    // C/D layout: VGPR r -> M = r (lanes 0-15) or r+8 (lanes 16-31); N = lane%16.
    const int rowBase0 = mBase + (lane >> 4) * 8;
    #pragma unroll
    for (int j = 0; j < 2; ++j) {
        const int col = nTile + nCol + 16 * j;
        float db = 0.0f;
        if (col < NODES) {
            const float2 b2 = *(const float2*)(bias + (size_t)col * 2);
            db = b2.y - b2.x;
        }
        #pragma unroll
        for (int r = 0; r < 8; ++r) {
            float d0 = acc[2 * j + 0][r] + db;
            float d1 = acc[2 * j + 1][r] + db;
            s[(size_t)(rowBase0 + r) * NPAD + col]      = 1.0f / (1.0f + __expf(-d0));
            s[(size_t)(rowBase0 + 16 + r) * NPAD + col] = 1.0f / (1.0f + __expf(-d1));
        }
    }
}

// ---------------------------------------------------------------------------
// Kernel 3: leaf-path products, 4 consecutive leaves per thread.
// For a quad-aligned leaf base, the 4 leaves share the top 12 path levels:
//   out[i=2j+m] = pc * (j ? sv2 : 1-sv2) * (m ? sv(A/B) : 1-sv(A/B))
// 15 gathers produce 4 outputs (vs 56), written as one float4.
// ---------------------------------------------------------------------------
__global__ __launch_bounds__(256) void leaf_prod_kernel(
        const float* __restrict__ s, float* __restrict__ out) {
    const int t    = blockIdx.x * 256 + threadIdx.x;     // B*V/4 threads
    const int b    = t >> 12;                            // 4096 quads per row
    const int quad = t & 4095;
    const unsigned hbase = (unsigned)(quad << 2) + VLEAF;   // leaf0 + V (mult of 4)
    const float* __restrict__ srow = s + (size_t)b * NPAD;

    // Common prefix: walk 12 levels starting from hbase>>2.
    float pc = 1.0f;
    unsigned h = hbase >> 2;
    #pragma unroll
    for (int d = 0; d < DEPTH - 2; ++d) {
        const unsigned bit = h & 1u;
        h >>= 1;
        const float sv = srow[h - 1];
        pc *= bit ? sv : (1.0f - sv);
    }
    // Level D-2: shared node, bit = j (hbase>>1 is even).
    const float sv2 = srow[(hbase >> 2) - 1];
    // Level D-1: node (hbase>>1)-1 for leaves {0,1}, (hbase>>1) for {2,3}; bit = i&1.
    const float svA = srow[(hbase >> 1) - 1];
    const float svB = srow[(hbase >> 1)];

    const float p0 = pc * (1.0f - sv2);
    const float p1 = pc * sv2;
    float4 o;
    o.x = p0 * (1.0f - svA);
    o.y = p0 * svA;
    o.z = p1 * (1.0f - svB);
    o.w = p1 * svB;
    *(float4*)(out + (size_t)b * VLEAF + (quad << 2)) = o;
}

// ---------------------------------------------------------------------------
// Launch
// ---------------------------------------------------------------------------
extern "C" void kernel_launch(void* const* d_in, const int* in_sizes, int n_in,
                              void* d_out, int out_size, void* d_ws, size_t ws_size,
                              hipStream_t stream) {
    const float* x    = (const float*)d_in[0];   // (256, 512)
    const float* W    = (const float*)d_in[1];   // (16383, 512, 2)
    const float* bias = (const float*)d_in[2];   // (16383, 2)
    // d_in[3]/d_in[4] = path_nodes/path_bits: recomputed arithmetically.
    float* out = (float*)d_out;                  // (256, 16384)

    // Workspace layout: [ s : 256*16384*4 = 16 MiB ][ xh : 256*512*2 = 256 KiB ]
    float*    s  = (float*)d_ws;
    _Float16* xh = (_Float16*)((char*)d_ws + (size_t)BATCH * NPAD * sizeof(float));

    cvt_x_kernel<<<(BATCH * EDIM) / 256, 256, 0, stream>>>(x, xh);
    gemm_sigmoid_kernel<<<NPAD / NT, 256, 0, stream>>>(W, bias, xh, s);
    leaf_prod_kernel<<<(BATCH * VLEAF / 4) / 256, 256, 0, stream>>>(s, out);
}
